// PointNet2_87729001988965
// MI455X (gfx1250) — matrix operand order repaired and checked by reading, not compiled, not run on hardware
//
#include <hip/hip_runtime.h>
#include <hip/hip_bf16.h>
#include <stdint.h>

// PointNet++ segmentation forward for MI455X (gfx1250, wave32, WMMA).
// All MLP layers run through one bf16 WMMA GEMM kernel (v_wmma_f32_16x16x32_bf16)
// with BN+bias+ReLU folded into the epilogue. Activations: (row=point, col=channel)
// bf16, channel dim padded to multiple of 32.

#define BQ   4
#define NPTS 16384
#define NCLS 8

typedef __attribute__((ext_vector_type(16))) __bf16 v16bf;
typedef __attribute__((ext_vector_type(8)))  float  v8f;

union FragBF {
  v16bf v;
  unsigned int u32[8];
  unsigned short u16[16];
};

__device__ __forceinline__ unsigned short f2bf(float f) {
  unsigned int u = __float_as_uint(f);
  unsigned int r = (u + 0x7FFFu + ((u >> 16) & 1u)) >> 16;  // RNE
  return (unsigned short)r;
}
__device__ __forceinline__ float bf2f(unsigned short h) {
  return __uint_as_float(((unsigned int)h) << 16);
}

// A fragment (16x32 bf16): lane (r,h) holds row co0+r,
//   VGPR j<4 : K = h*8 + 2j {lo,hi};  VGPR j>=4 : K = 16 + h*8 + 2(j-4).
__device__ __forceinline__ void load_a(FragBF& f, const unsigned short* wrow,
                                       int k0, int h) {
  const unsigned short* wp = wrow + k0 + h * 8;
#pragma unroll
  for (int j = 0; j < 4; ++j) f.u32[j]     = *(const unsigned int*)(wp + 2 * j);
#pragma unroll
  for (int j = 0; j < 4; ++j) f.u32[4 + j] = *(const unsigned int*)(wp + 16 + 2 * j);
}

// B fragment (32x16 bf16): lane (r,h) holds col m+r,
//   VGPR j<4 : K = h*8 + 2j {lo,hi};  VGPR j>=4 : K = 16 + h*8 + 2(j-4).
__device__ __forceinline__ void load_b(FragBF& f, const unsigned short* xrow,
                                       int k0, int h) {
  const unsigned short* xp = xrow + k0 + h * 8;
#pragma unroll
  for (int j = 0; j < 4; ++j) f.u32[j]     = *(const unsigned int*)(xp + 2 * j);
#pragma unroll
  for (int j = 0; j < 4; ++j) f.u32[4 + j] = *(const unsigned int*)(xp + 16 + 2 * j);
}

// ---------------------------------------------------------------------------
// WMMA GEMM + folded BN + ReLU.
//   Y[m, co] = relu( alpha[co] * sum_k W[co,k]*X[m,k] + (b[co]*alpha[co]+beta[co]) )
//   X: (M, Kp) bf16 bits, W: (Coutp, Kp) bf16 bits, Y: (M, Coutp) bf16 bits.
//
// grid = (M/64, Coutp/64), block = 128 (4 waves).
// Each wave computes a 16(cout) x 64(m) tile: ONE A (weight) fragment reused
// against FOUR B fragments -> 4 independent v_wmma per K-step.
// Explicit ping-pong software pipeline (named buffers, manual 2x unroll; no
// arrays so every fragment is register-resident): one buffer's loads are in
// flight while the other buffer's 4 WMMAs issue.
// ---------------------------------------------------------------------------
__global__ void __launch_bounds__(128) gemm_bn_relu_kernel(
    const unsigned short* __restrict__ X,
    const unsigned short* __restrict__ W,
    const float* __restrict__ gvec, const float* __restrict__ bvec,
    const float* __restrict__ betav,
    unsigned short* __restrict__ Y,
    int Kp, int Coutp)
{
  const int lane = threadIdx.x & 31;
  const int wave = threadIdx.x >> 5;
  const int r = lane & 15;
  const int h = lane >> 4;
  const int m0  = blockIdx.x * 64;
  const int co0 = blockIdx.y * 64 + wave * 16;

  const unsigned short* wrow  = W + (size_t)(co0 + r) * Kp;
  const unsigned short* xrow0 = X + (size_t)(m0 +  0 + r) * Kp;
  const unsigned short* xrow1 = X + (size_t)(m0 + 16 + r) * Kp;
  const unsigned short* xrow2 = X + (size_t)(m0 + 32 + r) * Kp;
  const unsigned short* xrow3 = X + (size_t)(m0 + 48 + r) * Kp;

  v8f acc0 = {}, acc1 = {}, acc2 = {}, acc3 = {};

  FragBF aA, p0A, p1A, p2A, p3A;   // ping buffer
  FragBF aB, p0B, p1B, p2B, p3B;   // pong buffer

  const int nsteps = Kp >> 5;

  // Preload step 0 into buffer A.
  load_a(aA, wrow, 0, h);
  load_b(p0A, xrow0, 0, h);
  load_b(p1A, xrow1, 0, h);
  load_b(p2A, xrow2, 0, h);
  load_b(p3A, xrow3, 0, h);

  int s_ = 0;
  for (; s_ + 2 <= nsteps; s_ += 2) {
    // Issue loads for step s+1 into buffer B, then consume buffer A.
    const int k1 = (s_ + 1) << 5;
    load_a(aB, wrow, k1, h);
    load_b(p0B, xrow0, k1, h);
    load_b(p1B, xrow1, k1, h);
    load_b(p2B, xrow2, k1, h);
    load_b(p3B, xrow3, k1, h);
    __builtin_prefetch(xrow0 + k1 + 32, 0, 1);
    __builtin_prefetch(xrow1 + k1 + 32, 0, 1);
    __builtin_prefetch(xrow2 + k1 + 32, 0, 1);
    __builtin_prefetch(xrow3 + k1 + 32, 0, 1);
    acc0 = __builtin_amdgcn_wmma_f32_16x16x32_bf16(false, aA.v, false, p0A.v,
                                                   (short)0, acc0, false, false);
    acc1 = __builtin_amdgcn_wmma_f32_16x16x32_bf16(false, aA.v, false, p1A.v,
                                                   (short)0, acc1, false, false);
    acc2 = __builtin_amdgcn_wmma_f32_16x16x32_bf16(false, aA.v, false, p2A.v,
                                                   (short)0, acc2, false, false);
    acc3 = __builtin_amdgcn_wmma_f32_16x16x32_bf16(false, aA.v, false, p3A.v,
                                                   (short)0, acc3, false, false);
    // Issue loads for step s+2 into buffer A, then consume buffer B.
    if (s_ + 2 < nsteps) {
      const int k2 = (s_ + 2) << 5;
      load_a(aA, wrow, k2, h);
      load_b(p0A, xrow0, k2, h);
      load_b(p1A, xrow1, k2, h);
      load_b(p2A, xrow2, k2, h);
      load_b(p3A, xrow3, k2, h);
    }
    acc0 = __builtin_amdgcn_wmma_f32_16x16x32_bf16(false, aB.v, false, p0B.v,
                                                   (short)0, acc0, false, false);
    acc1 = __builtin_amdgcn_wmma_f32_16x16x32_bf16(false, aB.v, false, p1B.v,
                                                   (short)0, acc1, false, false);
    acc2 = __builtin_amdgcn_wmma_f32_16x16x32_bf16(false, aB.v, false, p2B.v,
                                                   (short)0, acc2, false, false);
    acc3 = __builtin_amdgcn_wmma_f32_16x16x32_bf16(false, aB.v, false, p3B.v,
                                                   (short)0, acc3, false, false);
  }
  if (s_ < nsteps) {  // odd tail: buffer A holds the final K-step
    acc0 = __builtin_amdgcn_wmma_f32_16x16x32_bf16(false, aA.v, false, p0A.v,
                                                   (short)0, acc0, false, false);
    acc1 = __builtin_amdgcn_wmma_f32_16x16x32_bf16(false, aA.v, false, p1A.v,
                                                   (short)0, acc1, false, false);
    acc2 = __builtin_amdgcn_wmma_f32_16x16x32_bf16(false, aA.v, false, p2A.v,
                                                   (short)0, acc2, false, false);
    acc3 = __builtin_amdgcn_wmma_f32_16x16x32_bf16(false, aA.v, false, p3A.v,
                                                   (short)0, acc3, false, false);
  }

  // Per-channel folded BN constants (co depends on j only, shared by 4 tiles).
  const float s = rsqrtf(1.0f + 1e-5f);
  float alpha[8], bias[8];
  const int cobase = co0 + h * 8;
#pragma unroll
  for (int j = 0; j < 8; ++j) {
    float a_ = gvec[cobase + j] * s;
    alpha[j] = a_;
    bias[j] = bvec[cobase + j] * a_ + betav[cobase + j];
  }

#define STORE_TILE(AV, T)                                          \
  {                                                                \
    const int m = m0 + 16 * (T) + r;                               \
    unsigned short* yrow = Y + (size_t)m * Coutp + cobase;         \
    _Pragma("unroll")                                              \
    for (int j = 0; j < 8; ++j) {                                  \
      float v = (AV)[j] * alpha[j] + bias[j];                      \
      v = fmaxf(v, 0.0f);                                          \
      yrow[j] = f2bf(v);                                           \
    }                                                              \
  }
  STORE_TILE(acc0, 0)
  STORE_TILE(acc1, 1)
  STORE_TILE(acc2, 2)
  STORE_TILE(acc3, 3)
#undef STORE_TILE
}

// ---------------------------------------------------------------------------
// Farthest point sampling: one block per batch, dist[] in global scratch,
// block-wide argmax reduction in LDS.
// ---------------------------------------------------------------------------
__global__ void fps_kernel(const float* __restrict__ xyz, int Nn, int npoint,
                           int* __restrict__ out_idx, float* __restrict__ distbuf)
{
  const int b = blockIdx.x;
  const float* xb = xyz + (size_t)b * Nn * 3;
  float* dist = distbuf + (size_t)b * Nn;
  int* oi = out_idx + b * npoint;
  __shared__ float sv[256];
  __shared__ int   si[256];
  __shared__ int   s_far;
  const int tid = threadIdx.x;
  for (int i = tid; i < Nn; i += 256) dist[i] = 1e10f;
  if (tid == 0) s_far = 0;
  __syncthreads();
  for (int it = 0; it < npoint; ++it) {
    const int far = s_far;
    if (tid == 0) oi[it] = far;
    const float cx = xb[far * 3 + 0], cy = xb[far * 3 + 1], cz = xb[far * 3 + 2];
    float best = -1.0f; int bi = 0;
    for (int i = tid; i < Nn; i += 256) {
      float dx = xb[i * 3 + 0] - cx;
      float dy = xb[i * 3 + 1] - cy;
      float dz = xb[i * 3 + 2] - cz;
      float d = dx * dx + dy * dy + dz * dz;
      float dm = fminf(dist[i], d);
      dist[i] = dm;
      if (dm > best) { best = dm; bi = i; }
    }
    sv[tid] = best; si[tid] = bi;
    __syncthreads();
    for (int s = 128; s > 0; s >>= 1) {
      if (tid < s) {
        if (sv[tid + s] > sv[tid] ||
            (sv[tid + s] == sv[tid] && si[tid + s] < si[tid])) {
          sv[tid] = sv[tid + s]; si[tid] = si[tid + s];
        }
      }
      __syncthreads();
    }
    if (tid == 0) s_far = si[0];
    __syncthreads();
  }
}

__global__ void gather_xyz_kernel(const float* __restrict__ xyz,
                                  const int* __restrict__ idx,
                                  int Nn, int S, float* __restrict__ out)
{
  int t = blockIdx.x * blockDim.x + threadIdx.x;
  if (t >= BQ * S) return;
  int b = t / S;
  int n = idx[t];
  const float* p = xyz + ((size_t)b * Nn + n) * 3;
  out[(size_t)t * 3 + 0] = p[0];
  out[(size_t)t * 3 + 1] = p[1];
  out[(size_t)t * 3 + 2] = p[2];
}

// First nsample(=32) in-range neighbors in index order (matches sorted select).
__global__ void ball_query_kernel(const float* __restrict__ xyz,
                                  const float* __restrict__ new_xyz,
                                  int Nn, int S, float r2, int* __restrict__ gidx)
{
  int t = blockIdx.x * blockDim.x + threadIdx.x;
  if (t >= BQ * S) return;
  int b = t / S;
  const float* xb = xyz + (size_t)b * Nn * 3;
  float qx = new_xyz[(size_t)t * 3 + 0];
  float qy = new_xyz[(size_t)t * 3 + 1];
  float qz = new_xyz[(size_t)t * 3 + 2];
  int cnt = 0, first = -1;
  const size_t base = (size_t)t * 32;
  for (int n = 0; n < Nn && cnt < 32; ++n) {
    float dx = xb[n * 3 + 0] - qx;
    float dy = xb[n * 3 + 1] - qy;
    float dz = xb[n * 3 + 2] - qz;
    float d = dx * dx + dy * dy + dz * dz;
    if (d <= r2) {
      if (first < 0) first = n;
      gidx[base + cnt] = n;
      ++cnt;
    }
  }
  if (first < 0) first = 0;
  for (; cnt < 32; ++cnt) gidx[base + cnt] = first;
}

// Build grouped features: rows (b,s,kk), cols [xyz-center(3) | feat(C) | zero pad].
__global__ void group_kernel(const float* __restrict__ xyz, int Nn,
                             const float* __restrict__ new_xyz, int S,
                             const int* __restrict__ gidx,
                             const unsigned short* __restrict__ feat, int Cs, int C,
                             int Kp, unsigned short* __restrict__ Xg)
{
  int t = blockIdx.x * blockDim.x + threadIdx.x;
  if (t >= BQ * S * 32) return;
  int bs = t >> 5;
  int b = bs / S;
  int n = gidx[t];
  const float* p = xyz + ((size_t)b * Nn + n) * 3;
  const float* q = new_xyz + (size_t)bs * 3;
  unsigned short* o = Xg + (size_t)t * Kp;
  o[0] = f2bf(p[0] - q[0]);
  o[1] = f2bf(p[1] - q[1]);
  o[2] = f2bf(p[2] - q[2]);
  const unsigned short* f = feat + ((size_t)b * Nn + n) * Cs;
  for (int c = 0; c < C; ++c) o[3 + c] = f[c];
  for (int c = 3 + C; c < Kp; ++c) o[c] = 0;
}

__global__ void maxpool_kernel(const unsigned short* __restrict__ Y, int BS, int Cp,
                               unsigned short* __restrict__ out)
{
  int t = blockIdx.x * blockDim.x + threadIdx.x;
  if (t >= BS * Cp) return;
  int bs = t / Cp, c = t % Cp;
  const unsigned short* p = Y + ((size_t)bs * 32) * Cp + c;
  float m = -3.4e38f;
  for (int kk = 0; kk < 32; ++kk) m = fmaxf(m, bf2f(p[(size_t)kk * Cp]));
  out[t] = f2bf(m);
}

// 3-NN inverse-distance interpolation + concat [feat1(C1) | interp(C2) | pad].
__global__ void fp_interp_kernel(const float* __restrict__ xyz1, int N1,
                                 const float* __restrict__ xyz2, int N2,
                                 const unsigned short* __restrict__ feat1, int C1s, int C1,
                                 const unsigned short* __restrict__ feat2, int C2s, int C2,
                                 int Kp, unsigned short* __restrict__ Xo)
{
  int t = blockIdx.x * blockDim.x + threadIdx.x;
  if (t >= BQ * N1) return;
  int b = t / N1;
  const float* p = xyz1 + (size_t)t * 3;
  const float px = p[0], py = p[1], pz = p[2];
  const float* q = xyz2 + (size_t)b * N2 * 3;
  float d0 = 3.4e38f, d1 = 3.4e38f, d2 = 3.4e38f;
  int i0 = 0, i1 = 0, i2 = 0;
  for (int n = 0; n < N2; ++n) {
    float dx = q[n * 3 + 0] - px;
    float dy = q[n * 3 + 1] - py;
    float dz = q[n * 3 + 2] - pz;
    float d = dx * dx + dy * dy + dz * dz;
    if (d < d0)      { d2 = d1; i2 = i1; d1 = d0; i1 = i0; d0 = d; i0 = n; }
    else if (d < d1) { d2 = d1; i2 = i1; d1 = d; i1 = n; }
    else if (d < d2) { d2 = d; i2 = n; }
  }
  float r0 = 1.0f / (d0 + 1e-8f);
  float r1 = 1.0f / (d1 + 1e-8f);
  float r2 = 1.0f / (d2 + 1e-8f);
  float rs = r0 + r1 + r2;
  r0 /= rs; r1 /= rs; r2 /= rs;
  unsigned short* o = Xo + (size_t)t * Kp;
  const unsigned short* f1 = feat1 + (size_t)t * C1s;
  for (int c = 0; c < C1; ++c) o[c] = f1[c];
  const unsigned short* fb = feat2 + (size_t)b * N2 * C2s;
  const unsigned short* g0 = fb + (size_t)i0 * C2s;
  const unsigned short* g1 = fb + (size_t)i1 * C2s;
  const unsigned short* g2 = fb + (size_t)i2 * C2s;
  for (int c = 0; c < C2; ++c) {
    float v = r0 * bf2f(g0[c]) + r1 * bf2f(g1[c]) + r2 * bf2f(g2[c]);
    o[C1 + c] = f2bf(v);
  }
  for (int c = C1 + C2; c < Kp; ++c) o[c] = 0;
}

__global__ void f32_to_bf16_kernel(const float* __restrict__ in,
                                   unsigned short* __restrict__ out, int n)
{
  int t = blockIdx.x * blockDim.x + threadIdx.x;
  if (t < n) out[t] = f2bf(in[t]);
}

// Pack f32 weights (Cout, Cin) -> bf16 (Cout, Kp) with zero K-padding.
__global__ void pack_w_kernel(const float* __restrict__ w, int Cout, int Cin, int Kp,
                              unsigned short* __restrict__ out)
{
  int t = blockIdx.x * blockDim.x + threadIdx.x;
  if (t >= Cout * Kp) return;
  int co = t / Kp, k = t % Kp;
  out[t] = (k < Cin) ? f2bf(w[(size_t)co * Cin + k]) : (unsigned short)0;
}

// Final classifier: 128 -> 8, no BN/ReLU. out layout (B, 8, N) f32.
__global__ void head_final_kernel(const unsigned short* __restrict__ feat,
                                  const float* __restrict__ w2,
                                  const float* __restrict__ b2,
                                  float* __restrict__ out)
{
  int t = blockIdx.x * blockDim.x + threadIdx.x;
  if (t >= BQ * NPTS) return;
  int b = t / NPTS, n = t % NPTS;
  const unsigned short* f = feat + (size_t)t * 128;
  float acc[NCLS];
#pragma unroll
  for (int o = 0; o < NCLS; ++o) acc[o] = b2[o];
  for (int c = 0; c < 128; ++c) {
    float x = bf2f(f[c]);
#pragma unroll
    for (int o = 0; o < NCLS; ++o) acc[o] += x * w2[o * 128 + c];
  }
#pragma unroll
  for (int o = 0; o < NCLS; ++o)
    out[((size_t)b * NCLS + o) * NPTS + n] = acc[o];
}

// ---------------------------------------------------------------------------
extern "C" void kernel_launch(void* const* d_in, const int* in_sizes, int n_in,
                              void* d_out, int out_size, void* d_ws, size_t ws_size,
                              hipStream_t stream)
{
  (void)in_sizes; (void)n_in; (void)out_size; (void)ws_size;
  const int N = NPTS;
  const int S1 = 1024, S2 = 256, S3 = 64;
  const int M0 = BQ * N, M1 = BQ * S1, M2 = BQ * S2, M3 = BQ * S3;
  const int Mg1 = M1 * 32, Mg2 = M2 * 32, Mg3 = M3 * 32;

  // Input order assumption: setup_inputs() insertion order:
  // xyz, points, then params sa1,sa2,sa3,fp3,fp2,fp1 (each layer w,b,g,beta),
  // head (w1,b1,g1,beta1,w2,b2).
  const float* xyz0 = (const float*)d_in[0];
  const float* pts0 = (const float*)d_in[1];

  struct Layer { const float *w, *b, *g, *beta; int cout, cin, kp; unsigned short* wp; };
  int ci = 2;
  auto nxt = [&]() { return (const float*)d_in[ci++]; };
  auto round32 = [](int x) { return (x + 31) & ~31; };
  auto mk = [&](int cout, int cin) {
    Layer L; L.w = nxt(); L.b = nxt(); L.g = nxt(); L.beta = nxt();
    L.cout = cout; L.cin = cin; L.kp = round32(cin); L.wp = nullptr;
    return L;
  };
  Layer sa1[3] = { mk(64, 6),    mk(64, 64),   mk(128, 64)  };
  Layer sa2[3] = { mk(128, 131), mk(128, 128), mk(256, 128) };
  Layer sa3[3] = { mk(256, 259), mk(256, 256), mk(512, 256) };
  Layer fp3[2] = { mk(256, 768), mk(256, 256) };
  Layer fp2[2] = { mk(256, 384), mk(128, 256) };
  Layer fp1[3] = { mk(128, 131), mk(128, 128), mk(128, 128) };
  const float* hw1 = nxt(); const float* hb1 = nxt();
  const float* hg1 = nxt(); const float* hbt = nxt();
  const float* hw2 = nxt(); const float* hb2 = nxt();
  Layer head1 = { hw1, hb1, hg1, hbt, 128, 128, 128, nullptr };

  // ---- workspace layout ----
  char* wsp = (char*)d_ws; size_t off = 0;
  auto walloc = [&](size_t bytes) -> void* {
    void* p = wsp + off;
    off = (off + bytes + 255) & ~(size_t)255;
    return p;
  };
  unsigned short* PA = (unsigned short*)walloc((size_t)Mg1 * 128 * 2);  // 32 MB
  unsigned short* PB = (unsigned short*)walloc((size_t)Mg1 * 128 * 2);  // 32 MB
  unsigned short* ptsbf   = (unsigned short*)walloc((size_t)M0 * 3 * 2);
  unsigned short* featL1  = (unsigned short*)walloc((size_t)M1 * 128 * 2);
  unsigned short* featL2  = (unsigned short*)walloc((size_t)M2 * 256 * 2);
  unsigned short* featL3  = (unsigned short*)walloc((size_t)M3 * 512 * 2);
  unsigned short* featL2b = (unsigned short*)walloc((size_t)M2 * 256 * 2);
  unsigned short* featL1b = (unsigned short*)walloc((size_t)M1 * 128 * 2);
  float* xyz1 = (float*)walloc((size_t)M1 * 3 * 4);
  float* xyz2 = (float*)walloc((size_t)M2 * 3 * 4);
  float* xyz3 = (float*)walloc((size_t)M3 * 3 * 4);
  int* fi1 = (int*)walloc((size_t)M1 * 4);
  int* fi2 = (int*)walloc((size_t)M2 * 4);
  int* fi3 = (int*)walloc((size_t)M3 * 4);
  int* gq1 = (int*)walloc((size_t)Mg1 * 4);
  int* gq2 = (int*)walloc((size_t)Mg2 * 4);
  int* gq3 = (int*)walloc((size_t)Mg3 * 4);
  float* dist = (float*)walloc((size_t)M0 * 4);

  auto packL = [&](Layer& L) {
    L.wp = (unsigned short*)walloc((size_t)L.cout * L.kp * 2);
    int n = L.cout * L.kp;
    pack_w_kernel<<<(n + 255) / 256, 256, 0, stream>>>(L.w, L.cout, L.cin, L.kp, L.wp);
  };
  for (int i = 0; i < 3; ++i) packL(sa1[i]);
  for (int i = 0; i < 3; ++i) packL(sa2[i]);
  for (int i = 0; i < 3; ++i) packL(sa3[i]);
  for (int i = 0; i < 2; ++i) packL(fp3[i]);
  for (int i = 0; i < 2; ++i) packL(fp2[i]);
  for (int i = 0; i < 3; ++i) packL(fp1[i]);
  packL(head1);

  auto gemm = [&](const unsigned short* Xin, unsigned short* Yout, int M, Layer& L) {
    dim3 grid(M / 64, L.cout / 64);
    gemm_bn_relu_kernel<<<grid, 128, 0, stream>>>(Xin, L.wp, L.g, L.b, L.beta,
                                                  Yout, L.kp, L.cout);
  };

  // level-0 point features -> bf16
  f32_to_bf16_kernel<<<(M0 * 3 + 255) / 256, 256, 0, stream>>>(pts0, ptsbf, M0 * 3);

  // ---- SA1 ----
  fps_kernel<<<BQ, 256, 0, stream>>>(xyz0, N, S1, fi1, dist);
  gather_xyz_kernel<<<(M1 + 255) / 256, 256, 0, stream>>>(xyz0, fi1, N, S1, xyz1);
  ball_query_kernel<<<(M1 + 255) / 256, 256, 0, stream>>>(xyz0, xyz1, N, S1, 0.01f, gq1);
  group_kernel<<<(Mg1 + 255) / 256, 256, 0, stream>>>(xyz0, N, xyz1, S1, gq1,
                                                      ptsbf, 3, 3, 32, PA);
  gemm(PA, PB, Mg1, sa1[0]);
  gemm(PB, PA, Mg1, sa1[1]);
  gemm(PA, PB, Mg1, sa1[2]);
  maxpool_kernel<<<(M1 * 128 + 255) / 256, 256, 0, stream>>>(PB, M1, 128, featL1);

  // ---- SA2 ----
  fps_kernel<<<BQ, 256, 0, stream>>>(xyz1, S1, S2, fi2, dist);
  gather_xyz_kernel<<<(M2 + 255) / 256, 256, 0, stream>>>(xyz1, fi2, S1, S2, xyz2);
  ball_query_kernel<<<(M2 + 255) / 256, 256, 0, stream>>>(xyz1, xyz2, S1, S2, 0.04f, gq2);
  group_kernel<<<(Mg2 + 255) / 256, 256, 0, stream>>>(xyz1, S1, xyz2, S2, gq2,
                                                      featL1, 128, 128, 160, PA);
  gemm(PA, PB, Mg2, sa2[0]);
  gemm(PB, PA, Mg2, sa2[1]);
  gemm(PA, PB, Mg2, sa2[2]);
  maxpool_kernel<<<(M2 * 256 + 255) / 256, 256, 0, stream>>>(PB, M2, 256, featL2);

  // ---- SA3 ----
  fps_kernel<<<BQ, 256, 0, stream>>>(xyz2, S2, S3, fi3, dist);
  gather_xyz_kernel<<<(M3 + 255) / 256, 256, 0, stream>>>(xyz2, fi3, S2, S3, xyz3);
  ball_query_kernel<<<(M3 + 255) / 256, 256, 0, stream>>>(xyz2, xyz3, S2, S3, 0.16f, gq3);
  group_kernel<<<(Mg3 + 255) / 256, 256, 0, stream>>>(xyz2, S2, xyz3, S3, gq3,
                                                      featL2, 256, 256, 288, PA);
  gemm(PA, PB, Mg3, sa3[0]);
  gemm(PB, PA, Mg3, sa3[1]);
  gemm(PA, PB, Mg3, sa3[2]);
  maxpool_kernel<<<(M3 * 512 + 255) / 256, 256, 0, stream>>>(PB, M3, 512, featL3);

  // ---- FP3: level3 -> level2 ----
  fp_interp_kernel<<<(M2 + 255) / 256, 256, 0, stream>>>(
      xyz2, S2, xyz3, S3, featL2, 256, 256, featL3, 512, 512, 768, PA);
  gemm(PA, PB, M2, fp3[0]);
  gemm(PB, featL2b, M2, fp3[1]);

  // ---- FP2: level2 -> level1 ----
  fp_interp_kernel<<<(M1 + 255) / 256, 256, 0, stream>>>(
      xyz1, S1, xyz2, S2, featL1, 128, 128, featL2b, 256, 256, 384, PA);
  gemm(PA, PB, M1, fp2[0]);
  gemm(PB, featL1b, M1, fp2[1]);

  // ---- FP1: level1 -> level0 ----
  fp_interp_kernel<<<(M0 + 255) / 256, 256, 0, stream>>>(
      xyz0, N, xyz1, S1, ptsbf, 3, 3, featL1b, 128, 128, 160, PA);
  gemm(PA, PB, M0, fp1[0]);
  gemm(PB, PA, M0, fp1[1]);
  gemm(PA, PB, M0, fp1[2]);

  // ---- Head ----
  gemm(PB, PA, M0, head1);
  head_final_kernel<<<(M0 + 255) / 256, 256, 0, stream>>>(PA, hw2, hb2, (float*)d_out);
}
